// QuantLlamaAttentionFused_55911884259549
// MI455X (gfx1250) — compile-verified, matrix operand order checked by reading, not run on hardware
//
#include <hip/hip_runtime.h>
#include <hip/hip_bf16.h>
#include <stdint.h>

// Problem constants (B=1)
#define S_LEN 2048
#define HID   4096
#define NHEAD 32
#define HD    128

typedef __attribute__((ext_vector_type(16))) __bf16 v16bf;
typedef __attribute__((ext_vector_type(8)))  float  v8f;
typedef __attribute__((ext_vector_type(4)))  int    v4i;

typedef v4i __attribute__((address_space(1))) as1_v4i;  // global
typedef v4i __attribute__((address_space(3))) as3_v4i;  // LDS

#if defined(__has_builtin)
#if __has_builtin(__builtin_amdgcn_global_load_async_to_lds_b128)
#define HAVE_ASYNC_LDS 1
#endif
#if __has_builtin(__builtin_amdgcn_s_wait_asynccnt)
#define HAVE_WAIT_ASYNC_BUILTIN 1
#endif
#endif

__device__ inline void wait_asynccnt0() {
#if defined(HAVE_WAIT_ASYNC_BUILTIN)
  __builtin_amdgcn_s_wait_asynccnt(0);
#elif defined(HAVE_ASYNC_LDS)
  asm volatile("s_wait_asynccnt 0x0" ::: "memory");
#endif
}

// ---------- bf16 helpers (RNE) ----------
__device__ inline uint16_t f32_to_bf16(float f) {
  uint32_t u = __float_as_uint(f);
  u += 0x7FFFu + ((u >> 16) & 1u);
  return (uint16_t)(u >> 16);
}
__device__ inline uint32_t pack_bf16x2(float a, float b) {
  return (uint32_t)f32_to_bf16(a) | ((uint32_t)f32_to_bf16(b) << 16);
}

// ---------- WMMA 16-bit operand fragment loader ----------
// Source is row-major bf16 (uint16) with 'stride' elements per row, rows
// striped lane&15, 16-bit A-matrix layout (ISA 7.12.2):
//   VGPR v holds K pair: k = k0 + (v<4?0:16) + 8*(lane>>4) + 2*(v&3)
// Works for A tiles (rows = M) and, with an N-major tile, for B (rows = N).
__device__ inline void load_frag16(const uint16_t* base, int stride, int k0, uint32_t u[8]) {
  const int lane = threadIdx.x & 31;
  const uint32_t* p = (const uint32_t*)(base + (size_t)(lane & 15) * stride);
  const int kh = (lane >> 4) * 8;
#pragma unroll
  for (int v = 0; v < 8; ++v) {
    const int k = k0 + ((v & 4) ? 16 : 0) + kh + (v & 3) * 2;
    u[v] = p[k >> 1];
  }
}

__device__ inline v8f wmma_bf16(const uint32_t a[8], const uint32_t b[8], v8f c) {
  union { v16bf v; uint32_t u[8]; } A, B;
#pragma unroll
  for (int i = 0; i < 8; ++i) { A.u[i] = a[i]; B.u[i] = b[i]; }
  return __builtin_amdgcn_wmma_f32_16x16x32_bf16(false, A.v, false, B.v,
                                                 (short)0, c, false, false);
}

// ============================================================
// GEMM: C[M,N] f32 = A[M,K] (f32 or bf16) x B[K,N] f32
// 64x64 block tile, 128 threads (4 waves), each wave 32x32 (2x2 WMMA tiles),
// K-step 32, double-buffered LDS, f32->bf16 conversion during LDS fill.
// ============================================================
template<bool A_BF16>
__global__ __launch_bounds__(128) void gemm_bf16_wmma(const void* __restrict__ Ap,
                                                      const float* __restrict__ B,
                                                      float* __restrict__ C,
                                                      int M, int N, int K) {
  __shared__ uint16_t lA[2][64 * 40];   // [m][k], stride 40
  __shared__ uint16_t lB[2][64 * 40];   // [n][k], stride 40 (N-major => k pairs contiguous)

  const int m0 = blockIdx.y * 64;
  const int n0 = blockIdx.x * 64;
  const int t  = threadIdx.x;
  const int w  = t >> 5, wm = w >> 1, wn = w & 1;
  const int lane = t & 31, lq = lane & 15, half = lane >> 4;

  v8f acc[2][2];
#pragma unroll
  for (int i = 0; i < 2; ++i)
#pragma unroll
    for (int j = 0; j < 2; ++j)
#pragma unroll
      for (int v = 0; v < 8; ++v) acc[i][j][v] = 0.0f;

  auto fill = [&](int buf, int k0) {
    // A tile 64x32
    {
      const int kr = (t & 7) * 4, r0 = t >> 3;
#pragma unroll
      for (int r = r0; r < 64; r += 16) {
        uint32_t w0, w1;
        if (A_BF16) {
          const uint32_t* src =
              (const uint32_t*)((const uint16_t*)Ap + (size_t)(m0 + r) * K + k0 + kr);
          w0 = src[0]; w1 = src[1];
        } else {
          const float4 f = *(const float4*)((const float*)Ap + (size_t)(m0 + r) * K + k0 + kr);
          w0 = pack_bf16x2(f.x, f.y); w1 = pack_bf16x2(f.z, f.w);
        }
        uint32_t* dst = (uint32_t*)(&lA[buf][r * 40 + kr]);
        dst[0] = w0; dst[1] = w1;
      }
    }
    // B tile 32x64, stored transposed -> lB[n][k]
    {
      const int n4 = (t & 15) * 4, kk = t >> 4;
#pragma unroll
      for (int k = kk; k < 32; k += 8) {
        const float4 f = *(const float4*)(B + (size_t)(k0 + k) * N + n0 + n4);
        lB[buf][(n4 + 0) * 40 + k] = f32_to_bf16(f.x);
        lB[buf][(n4 + 1) * 40 + k] = f32_to_bf16(f.y);
        lB[buf][(n4 + 2) * 40 + k] = f32_to_bf16(f.z);
        lB[buf][(n4 + 3) * 40 + k] = f32_to_bf16(f.w);
      }
    }
  };

  const int nk = K >> 5;
  fill(0, 0);
  __syncthreads();

  for (int kt = 0; kt < nk; ++kt) {
    const int cur = kt & 1;
    uint32_t af0[8], af1[8], bq0[8], bq1[8];
    load_frag16(&lA[cur][(wm * 32) * 40],      40, 0, af0);
    load_frag16(&lA[cur][(wm * 32 + 16) * 40], 40, 0, af1);
    load_frag16(&lB[cur][(wn * 32) * 40],      40, 0, bq0);
    load_frag16(&lB[cur][(wn * 32 + 16) * 40], 40, 0, bq1);
    acc[0][0] = wmma_bf16(af0, bq0, acc[0][0]);
    acc[0][1] = wmma_bf16(af0, bq1, acc[0][1]);
    acc[1][0] = wmma_bf16(af1, bq0, acc[1][0]);
    acc[1][1] = wmma_bf16(af1, bq1, acc[1][1]);
    if (kt + 1 < nk) fill(cur ^ 1, (kt + 1) << 5);
    __syncthreads();
  }

  // C layout: lane holds col lq, VGPR v holds row v + 8*half
#pragma unroll
  for (int i = 0; i < 2; ++i)
#pragma unroll
    for (int j = 0; j < 2; ++j)
#pragma unroll
      for (int v = 0; v < 8; ++v) {
        const int row = m0 + wm * 32 + i * 16 + v + 8 * half;
        const int col = n0 + wn * 32 + j * 16 + lq;
        C[(size_t)row * N + col] = acc[i][j][v];
      }
}

// ============================================================
// RoPE (f32 math) + pack Q/K/V to bf16 [NH][S][HD]
// xqkv: f32 [S][3*HID], col = which*HID + h*HD + d
// ============================================================
__global__ __launch_bounds__(256) void rope_pack(const float* __restrict__ xqkv,
                                                 uint16_t* __restrict__ Q,
                                                 uint16_t* __restrict__ Ko,
                                                 uint16_t* __restrict__ V) {
  const int idx = blockIdx.x * blockDim.x + threadIdx.x;  // S*NH*64
  if (idx >= S_LEN * NHEAD * 64) return;
  const int j = idx & 63;
  const int h = (idx >> 6) & (NHEAD - 1);
  const int s = idx >> 11;

  const float inv = __powf(10000.0f, -(float)(2 * j) / 128.0f);
  const float ang = (float)s * inv;
  float sn, cs;
  __sincosf(ang, &sn, &cs);

  const size_t base = (size_t)s * (3 * HID) + (size_t)h * HD;
  const float q1 = xqkv[base + j],            q2 = xqkv[base + j + 64];
  const float k1 = xqkv[base + HID + j],      k2 = xqkv[base + HID + j + 64];
  const float v1 = xqkv[base + 2 * HID + j],  v2 = xqkv[base + 2 * HID + j + 64];

  const size_t ob = ((size_t)h * S_LEN + s) * HD;
  Q[ob + j]       = f32_to_bf16(q1 * cs - q2 * sn);
  Q[ob + j + 64]  = f32_to_bf16(q2 * cs + q1 * sn);
  Ko[ob + j]      = f32_to_bf16(k1 * cs - k2 * sn);
  Ko[ob + j + 64] = f32_to_bf16(k2 * cs + k1 * sn);
  V[ob + j]       = f32_to_bf16(v1);
  V[ob + j + 64]  = f32_to_bf16(v2);
}

// ============================================================
// Flash attention (causal, online softmax), bf16 WMMA, fp32 accum.
// Grid: (S/64, NH), 128 threads = 4 waves; wave w owns q rows q0+16w..+15.
// K tile streamed with GLOBAL_LOAD_ASYNC_TO_LDS_B128 (ASYNCcnt); each wave
// drains its own ASYNCcnt before the workgroup barrier (the barrier orders
// execution, not other waves' outstanding DMAs).
// ============================================================
__global__ __launch_bounds__(128) void flash_attn(const uint16_t* __restrict__ Qb,
                                                  const uint16_t* __restrict__ Kb,
                                                  const uint16_t* __restrict__ Vb,
                                                  uint16_t* __restrict__ O) {
  __shared__ uint16_t lK[64 * 128];      // [key][hd]   (B-frag for Q*K^T)
  __shared__ uint16_t lVt[128 * 64];     // [hd][key]   (B-frag for P*V)
  __shared__ uint16_t lP[4][16 * 64];    // per-wave P staging [q][key]

  const int head = blockIdx.y;
  const int q0   = blockIdx.x * 64;
  const int t    = threadIdx.x;
  const int w    = t >> 5;
  const int lane = t & 31, lq = lane & 15, half = lane >> 4;
  const int qw   = q0 + w * 16;

  // Q fragments live in registers for the whole KV loop (HD=128 -> 4 chunks)
  const uint16_t* qbase = Qb + ((size_t)head * S_LEN + qw) * HD;
  uint32_t qa[4][8];
#pragma unroll
  for (int c = 0; c < 4; ++c) load_frag16(qbase, HD, c * 32, qa[c]);

  float m_i[8], l_i[8];
  v8f o[8];
#pragma unroll
  for (int v = 0; v < 8; ++v) { m_i[v] = -1e30f; l_i[v] = 0.0f; }
#pragma unroll
  for (int tn = 0; tn < 8; ++tn)
#pragma unroll
    for (int v = 0; v < 8; ++v) o[tn][v] = 0.0f;

  const float scale = 0.08838834764831845f;  // 1/sqrt(128)
  const int kv_end = q0 + 64;                // causal bound for this q block

  for (int kv0 = 0; kv0 < kv_end; kv0 += 64) {
    if (kv0 + 64 < kv_end) {  // CDNA5 global_prefetch_b8 on next K/V tiles
      __builtin_prefetch(Kb + ((size_t)head * S_LEN + kv0 + 64) * HD, 0, 1);
      __builtin_prefetch(Vb + ((size_t)head * S_LEN + kv0 + 64) * HD, 0, 1);
    }
    // K tile [64][128] straight copy
    {
      const uint16_t* src = Kb + ((size_t)head * S_LEN + kv0) * HD;
#if defined(HAVE_ASYNC_LDS)
      const int c8 = (t & 15) * 8;  // 16 bytes per lane along hd
#pragma unroll
      for (int r = t >> 4; r < 64; r += 8) {
        __builtin_amdgcn_global_load_async_to_lds_b128(
            (as1_v4i*)(src + (size_t)r * HD + c8),
            (as3_v4i*)(&lK[r * 128 + c8]),
            0, 0);
      }
#else
      const int c4 = (t & 31) * 4;
#pragma unroll
      for (int r = t >> 5; r < 64; r += 4) {
        const uint2 d = *(const uint2*)(src + (size_t)r * HD + c4);
        *(uint2*)(&lK[r * 128 + c4]) = d;
      }
#endif
    }
    // V tile transposed -> [hd][key] (needs per-element shuffle; VALU path)
    {
      const uint16_t* src = Vb + ((size_t)head * S_LEN + kv0) * HD;
      const int h4 = (t & 31) * 4;
#pragma unroll
      for (int r = t >> 5; r < 64; r += 4) {
        const uint2 d = *(const uint2*)(src + (size_t)r * HD + h4);
        lVt[(h4 + 0) * 64 + r] = (uint16_t)(d.x & 0xffff);
        lVt[(h4 + 1) * 64 + r] = (uint16_t)(d.x >> 16);
        lVt[(h4 + 2) * 64 + r] = (uint16_t)(d.y & 0xffff);
        lVt[(h4 + 3) * 64 + r] = (uint16_t)(d.y >> 16);
      }
    }
    wait_asynccnt0();   // drain this wave's async DMAs before the barrier
    __syncthreads();

    // scores: 4 subtiles of 16 keys, K-dim = HD via 4 chained WMMAs
    v8f sc[4];
#pragma unroll
    for (int sn = 0; sn < 4; ++sn) {
#pragma unroll
      for (int v = 0; v < 8; ++v) sc[sn][v] = 0.0f;
#pragma unroll
      for (int c = 0; c < 4; ++c) {
        uint32_t bk[8];
        load_frag16(&lK[(sn * 16) * 128], 128, c * 32, bk);
        sc[sn] = wmma_bf16(qa[c], bk, sc[sn]);
      }
      const int col = kv0 + sn * 16 + lq;
#pragma unroll
      for (int v = 0; v < 8; ++v) {
        const int row = qw + v + 8 * half;
        const float x = sc[sn][v] * scale;
        sc[sn][v] = (col <= row) ? x : -1e30f;
      }
    }

    // online softmax (row = VGPR index v; 16-lane reductions within half-wave)
    float corr[8];
#pragma unroll
    for (int v = 0; v < 8; ++v) {
      float mb = fmaxf(fmaxf(sc[0][v], sc[1][v]), fmaxf(sc[2][v], sc[3][v]));
#pragma unroll
      for (int off = 1; off < 16; off <<= 1)
        mb = fmaxf(mb, __shfl_xor(mb, off, 32));
      const float nm = fmaxf(m_i[v], mb);
      corr[v] = __expf(m_i[v] - nm);
      m_i[v] = nm;
    }
#pragma unroll
    for (int sn = 0; sn < 4; ++sn)
#pragma unroll
      for (int v = 0; v < 8; ++v)
        sc[sn][v] = __expf(sc[sn][v] - m_i[v]);
#pragma unroll
    for (int v = 0; v < 8; ++v) {
      float rs = sc[0][v] + sc[1][v] + sc[2][v] + sc[3][v];
#pragma unroll
      for (int off = 1; off < 16; off <<= 1)
        rs += __shfl_xor(rs, off, 32);
      l_i[v] = l_i[v] * corr[v] + rs;
    }
#pragma unroll
    for (int tn = 0; tn < 8; ++tn)
#pragma unroll
      for (int v = 0; v < 8; ++v) o[tn][v] *= corr[v];

    // P: C-layout -> A-layout via per-wave LDS (same-wave LDS ops are in-order)
#pragma unroll
    for (int sn = 0; sn < 4; ++sn)
#pragma unroll
      for (int v = 0; v < 8; ++v)
        lP[w][(v + 8 * half) * 64 + sn * 16 + lq] = f32_to_bf16(sc[sn][v]);

    // O += P * V  (kv chunks of 32, 8 hd tiles of 16)
#pragma unroll
    for (int c2 = 0; c2 < 2; ++c2) {
      uint32_t pa[8];
      load_frag16(lP[w], 64, c2 * 32, pa);
#pragma unroll
      for (int tn = 0; tn < 8; ++tn) {
        uint32_t bv[8];
        load_frag16(&lVt[(tn * 16) * 64], 64, c2 * 32, bv);
        o[tn] = wmma_bf16(pa, bv, o[tn]);
      }
    }
    __syncthreads();
  }

  // epilogue: O[s][h*HD + d] bf16, normalized
#pragma unroll
  for (int tn = 0; tn < 8; ++tn)
#pragma unroll
    for (int v = 0; v < 8; ++v) {
      const int row = qw + v + 8 * half;
      const int col = head * HD + tn * 16 + lq;
      O[(size_t)row * HID + col] = f32_to_bf16(o[tn][v] / l_i[v]);
    }
}

// ============================================================
extern "C" void kernel_launch(void* const* d_in, const int* in_sizes, int n_in,
                              void* d_out, int out_size, void* d_ws, size_t ws_size,
                              hipStream_t stream) {
  const float* x    = (const float*)d_in[0];  // [1,S,H]
  const float* Wqkv = (const float*)d_in[1];  // [H,3H]
  const float* Wo   = (const float*)d_in[2];  // [H,H]
  // d_in[3] position_ids / d_in[4] attention_mask: RoPE + causal mask computed analytically
  float* out = (float*)d_out;

  char* ws = (char*)d_ws;
  float* xqkv = (float*)ws;  // S x 3H f32 (96 MB)
  size_t off = (size_t)S_LEN * 3 * HID * sizeof(float);
  uint16_t* Qb = (uint16_t*)(ws + off); off += (size_t)NHEAD * S_LEN * HD * 2;
  uint16_t* Kb = (uint16_t*)(ws + off); off += (size_t)NHEAD * S_LEN * HD * 2;
  uint16_t* Vb = (uint16_t*)(ws + off); off += (size_t)NHEAD * S_LEN * HD * 2;
  uint16_t* Ab = (uint16_t*)(ws + off);  // attn out bf16 [S][H]

  gemm_bf16_wmma<false><<<dim3(3 * HID / 64, S_LEN / 64), 128, 0, stream>>>(
      x, Wqkv, xqkv, S_LEN, 3 * HID, HID);
  rope_pack<<<(S_LEN * NHEAD * 64) / 256, 256, 0, stream>>>(xqkv, Qb, Kb, Vb);
  flash_attn<<<dim3(S_LEN / 64, NHEAD), 128, 0, stream>>>(Qb, Kb, Vb, Ab);
  gemm_bf16_wmma<true><<<dim3(HID / 64, S_LEN / 64), 128, 0, stream>>>(
      Ab, Wo, out, S_LEN, HID, HID);
}
